// Linear_Attention_41034117546100
// MI455X (gfx1250) — matrix operand order repaired and checked by reading, not compiled, not run on hardware
//
#include <hip/hip_runtime.h>

// Linear attention memory (B=4,H=16,M=N=4096,D=V=128), f32 in/out,
// bf16 WMMA internally (memory-bound: ~26 GFLOP vs ~0.54 GB @ 23.3 TB/s).

#define BATCH 4
#define HEADS 16
#define MLEN  4096
#define NLEN  4096
#define DDIM  128
#define VDIM  128
#define BHN   (BATCH * HEADS)
#define MSTR  144      // padded LDS stride for the (M | Z | 0) tile: cols 0..127=M, 128=Z
#define NSPLIT 8       // N-dimension splits for the state-update reduction

typedef __attribute__((ext_vector_type(16))) __bf16 v16bf;
typedef __attribute__((ext_vector_type(8)))  float  v8f;

union Frag { v16bf v; unsigned long long d[4]; };

__device__ __forceinline__ unsigned short f2bf(float f) {
  unsigned int u = __float_as_uint(f);
  u += 0x7FFFu + ((u >> 16) & 1u);     // round-to-nearest-even
  return (unsigned short)(u >> 16);
}

__device__ __forceinline__ float sigma_f(float x) {   // elu(x)+1
  return x > 0.f ? x + 1.f : __expf(x);
}

__device__ __forceinline__ float beta_gate(const float* beta) {
  float b = 1.f / (1.f + __expf(-beta[0]));
  return fminf(0.999f, fmaxf(0.9f, b));
}

// ---- load helpers matching cdna5 WMMA VGPR layouts (16-bit operands) ----
// A-matrix 16x32: lane<16 holds K=[kb..kb+7],[kb+16..kb+23]; lane>=16 kb+=8.
__device__ __forceinline__ void loadA(Frag& f, const unsigned short* p) {
  f.d[0] = *(const unsigned long long*)(p);
  f.d[1] = *(const unsigned long long*)(p + 4);
  f.d[2] = *(const unsigned long long*)(p + 16);
  f.d[3] = *(const unsigned long long*)(p + 20);
}
// B-matrix 32x16: lane holds column n=l%16; K=[kb..kb+7],[kb+16..kb+23]; lane>=16 kb+=8.
__device__ __forceinline__ void loadB(Frag& f, const unsigned short* p, int ld) {
  unsigned short tmp[16];
#pragma unroll
  for (int s = 0; s < 16; ++s) tmp[s] = p[(s < 8 ? s : s + 8) * ld];
  __builtin_memcpy(&f, tmp, sizeof(tmp));
}

// ============================================================
// Kernel 1: A_mem = (sigma(q) @ M) / (sigma(q) @ Z)
// grid (MLEN/64, BH), block 128 (4 waves); wave owns a 16-row m-tile.
// ============================================================
__global__ __launch_bounds__(128)
void amem_kernel(const float* __restrict__ q, const float* __restrict__ Mm,
                 const float* __restrict__ Z, float* __restrict__ Aout) {
  __shared__ unsigned short Mlds[DDIM * MSTR];    // 36,864 B (M + Z col + zeros)
  __shared__ unsigned short sqlds[64 * DDIM];     // 16,384 B

  const int tid  = threadIdx.x;
  const int bh   = blockIdx.y;
  const int row0 = blockIdx.x * 64;

  const float* qg = q  + ((size_t)bh * MLEN + row0) * DDIM;
  const float* Mg = Mm + (size_t)bh * DDIM * VDIM;
  const float* Zg = Z  + (size_t)bh * DDIM;
  float*       Ag = Aout + ((size_t)bh * MLEN + row0) * VDIM;

  for (int i = tid; i < DDIM * VDIM; i += 128) {
    int d = i >> 7, c = i & 127;
    Mlds[d * MSTR + c] = f2bf(Mg[i]);
  }
  {
    int d = tid;                                   // block is exactly 128 threads
    Mlds[d * MSTR + 128] = f2bf(Zg[d]);
    for (int c = 129; c < MSTR; ++c) Mlds[d * MSTR + c] = 0;
  }
  for (int i = tid; i < 64 * DDIM; i += 128) {
    int r = i >> 7, c = i & 127;
    sqlds[r * DDIM + c] = f2bf(sigma_f(qg[(size_t)r * DDIM + c]));
  }
  __syncthreads();

  const int wave = tid >> 5, lane = tid & 31;
  const int laneM = lane & 15;
  const int hi = (lane >> 4) & 1;

  Frag af[4];
  const int arow = wave * 16 + laneM;
#pragma unroll
  for (int ks = 0; ks < 4; ++ks)
    loadA(af[ks], &sqlds[arow * DDIM + ks * 32 + hi * 8]);

  // denominator pass: B columns 128..143 (col 128 = Z)
  v8f accd = {};
#pragma unroll
  for (int ks = 0; ks < 4; ++ks) {
    Frag bf; loadB(bf, &Mlds[(ks * 32 + hi * 8) * MSTR + 128 + laneM], MSTR);
    accd = __builtin_amdgcn_wmma_f32_16x16x32_bf16(false, af[ks].v, false, bf.v,
                                                   (short)0, accd, false, false);
  }
  float rden[8];
#pragma unroll
  for (int r = 0; r < 8; ++r) {
    float dv = __shfl(accd[r], lane & 16, 32);     // C col 0 lives in lanes 0 / 16
    rden[r] = __builtin_amdgcn_rcpf(dv);           // den > 0 (sigma>0, Z>0)
  }

#pragma unroll
  for (int t = 0; t < 8; ++t) {
    v8f acc = {};
#pragma unroll
    for (int ks = 0; ks < 4; ++ks) {
      Frag bf; loadB(bf, &Mlds[(ks * 32 + hi * 8) * MSTR + t * 16 + laneM], MSTR);
      acc = __builtin_amdgcn_wmma_f32_16x16x32_bf16(false, af[ks].v, false, bf.v,
                                                    (short)0, acc, false, false);
    }
#pragma unroll
    for (int r = 0; r < 8; ++r) {
      int m = wave * 16 + r + hi * 8;
      Ag[(size_t)m * VDIM + t * 16 + laneM] = acc[r] * rden[r];
    }
  }
}

// ============================================================
// Kernel 2: M_new = b*M, Z_new = b*Z (init before atomic accumulation)
// ============================================================
__global__ __launch_bounds__(256)
void state_init_kernel(const float* __restrict__ Mm, const float* __restrict__ Z,
                       const float* __restrict__ beta,
                       float* __restrict__ Mout, float* __restrict__ Zout) {
  const float b = beta_gate(beta);
  size_t i = (size_t)blockIdx.x * blockDim.x + threadIdx.x;
  if (i < (size_t)BHN * DDIM * VDIM) Mout[i] = b * Mm[i];
  if (i < (size_t)BHN * DDIM)        Zout[i] = b * Z[i];
}

// ============================================================
// Kernel 3: M_new += sigma(k)^T @ (v - moment), Z_new += sum_n sigma(k)
//   moment = (sigma(k) @ M) / (sigma(k) @ Z)
// grid (BH, NSPLIT), block 256 (8 waves). Each WG streams 16 chunks of 32 rows.
// ============================================================
__global__ __launch_bounds__(256)
void mupdate_kernel(const float* __restrict__ k, const float* __restrict__ v,
                    const float* __restrict__ Mm, const float* __restrict__ Z,
                    float* __restrict__ Mout, float* __restrict__ Zout) {
  __shared__ unsigned short Mlds[DDIM * MSTR];   // 36,864 B
  __shared__ unsigned short sklds[32 * DDIM];    //  8,192 B  sigma(k) row-major
  __shared__ unsigned short skT[DDIM * 40];      // 10,240 B  sigma(k)^T (padded)
  __shared__ unsigned short ulds[32 * 136];      //  8,704 B  U = v - moment
  __shared__ float denLds[32];
  __shared__ float zsLds[256];

  const int tid = threadIdx.x;
  const int bh  = blockIdx.x;
  const int split = blockIdx.y;

  const float* kg = k + (size_t)bh * NLEN * DDIM;
  const float* vg = v + (size_t)bh * NLEN * VDIM;
  const float* Mg = Mm + (size_t)bh * DDIM * VDIM;
  const float* Zg = Z  + (size_t)bh * DDIM;
  float* Mo = Mout + (size_t)bh * DDIM * VDIM;
  float* Zo = Zout + (size_t)bh * DDIM;

  for (int i = tid; i < DDIM * VDIM; i += 256) {
    int d = i >> 7, c = i & 127;
    Mlds[d * MSTR + c] = f2bf(Mg[i]);
  }
  if (tid < DDIM) {
    Mlds[tid * MSTR + 128] = f2bf(Zg[tid]);
    for (int c = 129; c < MSTR; ++c) Mlds[tid * MSTR + c] = 0;
  }
  __syncthreads();

  const int wave = tid >> 5, lane = tid & 31;
  const int laneM = lane & 15;
  const int hi = (lane >> 4) & 1;
  const int dcol = tid & 127;     // this thread's fixed feature column
  const int nh = tid >> 7;        // 0..1

  float zacc = 0.f;
  v8f macc[8];
#pragma unroll
  for (int t = 0; t < 8; ++t)
#pragma unroll
    for (int e = 0; e < 8; ++e) macc[t][e] = 0.f;

  const int nchunks = (NLEN / 32) / NSPLIT;      // 16 chunks of 32 rows
  for (int cc = 0; cc < nchunks; ++cc) {
    const size_t n0 = ((size_t)split * nchunks + cc) * 32;

    // stage sigma(k) chunk (row-major + transposed); coalesced along d.
    // Prefetch the matching v rows (consumed next phase) into cache.
#pragma unroll
    for (int j = 0; j < 16; ++j) {
      int n = nh + 2 * j;
      __builtin_prefetch(&vg[(n0 + n) * VDIM + dcol], 0, 0);  // global_prefetch_b8
      float s = sigma_f(kg[(n0 + n) * DDIM + dcol]);
      zacc += s;
      unsigned short sb = f2bf(s);
      sklds[n * DDIM + dcol] = sb;
      skT[dcol * 40 + n] = sb;
    }
    __syncthreads();

    // Phase A (waves 0,1): denominators sigma(k)·Z via Z column of Mlds
    if (wave < 2) {
      const int mt = wave;
      v8f accd = {};
      const int arow = mt * 16 + laneM;
#pragma unroll
      for (int ks = 0; ks < 4; ++ks) {
        Frag af; loadA(af, &sklds[arow * DDIM + ks * 32 + hi * 8]);
        Frag bf; loadB(bf, &Mlds[(ks * 32 + hi * 8) * MSTR + 128 + laneM], MSTR);
        accd = __builtin_amdgcn_wmma_f32_16x16x32_bf16(false, af.v, false, bf.v,
                                                       (short)0, accd, false, false);
      }
      if (laneM == 0) {
#pragma unroll
        for (int r = 0; r < 8; ++r) denLds[mt * 16 + r + hi * 8] = accd[r];
      }
    }
    __syncthreads();

    // Phase B: moment numerator; U = v - num/den -> ulds (bf16)
    for (int jj = wave; jj < 16; jj += 8) {
      const int mt = jj >> 3, vt = jj & 7;
      v8f acc = {};
      const int arow = mt * 16 + laneM;
#pragma unroll
      for (int ks = 0; ks < 4; ++ks) {
        Frag af; loadA(af, &sklds[arow * DDIM + ks * 32 + hi * 8]);
        Frag bf; loadB(bf, &Mlds[(ks * 32 + hi * 8) * MSTR + vt * 16 + laneM], MSTR);
        acc = __builtin_amdgcn_wmma_f32_16x16x32_bf16(false, af.v, false, bf.v,
                                                      (short)0, acc, false, false);
      }
#pragma unroll
      for (int r = 0; r < 8; ++r) {
        int n = mt * 16 + r + hi * 8;
        int vcol = vt * 16 + laneM;
        float rd = __builtin_amdgcn_rcpf(denLds[n]);   // den > 0
        float uv = vg[(n0 + n) * VDIM + vcol] - acc[r] * rd;
        ulds[n * 136 + vcol] = f2bf(uv);
      }
    }
    __syncthreads();

    // Phase C: macc += sk^T(d-tile = wave, K=32) x U
    {
      Frag af2; loadA(af2, &skT[(wave * 16 + laneM) * 40 + hi * 8]);
#pragma unroll
      for (int vt = 0; vt < 8; ++vt) {
        Frag bf; loadB(bf, &ulds[(hi * 8) * 136 + vt * 16 + laneM], 136);
        macc[vt] = __builtin_amdgcn_wmma_f32_16x16x32_bf16(false, af2.v, false, bf.v,
                                                           (short)0, macc[vt], false, false);
      }
    }
    __syncthreads();
  }

  // accumulate this split's partial rank-512 update into M_new
#pragma unroll
  for (int vt = 0; vt < 8; ++vt)
#pragma unroll
    for (int r = 0; r < 8; ++r) {
      int d = wave * 16 + r + hi * 8;
      atomicAdd(&Mo[(size_t)d * VDIM + vt * 16 + laneM], macc[vt][r]);
    }

  zsLds[tid] = zacc;
  __syncthreads();
  if (tid < DDIM) atomicAdd(&Zo[tid], zsLds[tid] + zsLds[tid + 128]);
}

// ============================================================
extern "C" void kernel_launch(void* const* d_in, const int* in_sizes, int n_in,
                              void* d_out, int out_size, void* d_ws, size_t ws_size,
                              hipStream_t stream) {
  (void)in_sizes; (void)n_in; (void)d_ws; (void)ws_size; (void)out_size;
  const float* q    = (const float*)d_in[0];
  const float* kk   = (const float*)d_in[1];
  const float* v    = (const float*)d_in[2];
  const float* Mm   = (const float*)d_in[3];
  const float* Z    = (const float*)d_in[4];
  const float* beta = (const float*)d_in[5];

  float* out  = (float*)d_out;
  float* Aout = out;                                        // B,H,M,V
  float* Mout = Aout + (size_t)BHN * MLEN * VDIM;           // B,H,D,V
  float* Zout = Mout + (size_t)BHN * DDIM * VDIM;           // B,H,D,1

  amem_kernel<<<dim3(MLEN / 64, BHN), 128, 0, stream>>>(q, Mm, Z, Aout);
  state_init_kernel<<<(BHN * DDIM * VDIM + 255) / 256, 256, 0, stream>>>(Mm, Z, beta,
                                                                         Mout, Zout);
  mupdate_kernel<<<dim3(BHN, NSPLIT), 256, 0, stream>>>(kk, v, Mm, Z, Mout, Zout);
}
